// GraphAttentionNetwork_47888885350735
// MI455X (gfx1250) — matrix operand order repaired and checked by reading, not compiled, not run on hardware
//
#include <hip/hip_runtime.h>

// ---------------------------------------------------------------------------
// GAT (3 layers, 4 heads) for MI455X / gfx1250:
//   wave32 + v_wmma_f32_16x16x32_bf16 + double-buffered async global->LDS
//   staging (ASYNCcnt) in the dominant attention-aggregation kernel.
// ---------------------------------------------------------------------------

typedef __bf16 bf16_t;
typedef __attribute__((ext_vector_type(16))) __bf16 v16bf;
typedef __attribute__((ext_vector_type(8)))  __bf16 v8bf;
typedef __attribute__((ext_vector_type(2)))  __bf16 v2bf;
typedef __attribute__((ext_vector_type(8)))  float  v8f;
typedef __attribute__((ext_vector_type(2)))  float  v2f;

static constexpr int N      = 4096;
static constexpr int F      = 256;   // input width of every layer (F_IN == HID)
static constexpr int OUT    = 128;
static constexpr int HEADS  = 4;
static constexpr float ALPHA = 0.2f;

__device__ __forceinline__ bf16_t f2bf(float f) {
  unsigned u = __builtin_bit_cast(unsigned, f);
  u += 0x7fffu + ((u >> 16) & 1u);                 // round-to-nearest-even
  unsigned short h = (unsigned short)(u >> 16);
  return __builtin_bit_cast(bf16_t, h);
}
__device__ __forceinline__ float bf2f(bf16_t b) {
  unsigned u = ((unsigned)__builtin_bit_cast(unsigned short, b)) << 16;
  return __builtin_bit_cast(float, u);
}
// packed f32x2 -> bf16x2
__device__ __forceinline__ v2bf pk2bf(float a, float b) {
#if __has_builtin(__builtin_amdgcn_cvt_pk_bf16_f32)
  return __builtin_bit_cast(v2bf, __builtin_amdgcn_cvt_pk_bf16_f32(a, b));
#else
  v2f x = {a, b};
  return __builtin_convertvector(x, v2bf);
#endif
}

// async 16B global -> LDS copy, tracked by ASYNCcnt (ISA §10.6 / §15.18.3)
__device__ __forceinline__ void async_copy16(void* ldsdst, const void* gsrc) {
  unsigned l = (unsigned)(uintptr_t)ldsdst;        // low 32 bits == LDS offset
  asm volatile("global_load_async_to_lds_b128 %0, %1, off"
               :: "v"(l), "v"(gsrc) : "memory");
}
__device__ __forceinline__ void async_wait0() {
  asm volatile("s_wait_asynccnt 0" ::: "memory");
}

// ---------------------------------------------------------------------------
// 1) W [H][F][d] fp32  ->  WT [H][d][F] bf16 (transposed, B-fragment friendly)
// ---------------------------------------------------------------------------
__global__ void wt_kernel(const float* __restrict__ W, bf16_t* __restrict__ WT, int d) {
  int idx = blockIdx.x * 256 + threadIdx.x;        // over H*d*F
  if (idx >= HEADS * d * F) return;
  int f  = idx % F;
  int nd = (idx / F) % d;
  int h  = idx / (F * d);
  WT[idx] = f2bf(W[((size_t)h * F + f) * d + nd]);
}

// ---------------------------------------------------------------------------
// 2) Wh = x @ W (per head), stored transposed as bf16: WhT [H][d][N]
//    grid (N/64, H), block 128 (4 waves); wave w owns rows rb*64 + w*16..+16.
// ---------------------------------------------------------------------------
template <int DT>   // DT = d/16  (4 -> d=64, 8 -> d=128)
__global__ __launch_bounds__(128) void gemm_wh(const float* __restrict__ X,
                                               const bf16_t* __restrict__ WT,
                                               bf16_t* __restrict__ WhT) {
  __shared__ __align__(32) bf16_t sX[64 * 32];     // [64 rows][32 k] bf16
  const int h    = blockIdx.y;
  const int rb   = blockIdx.x * 64;
  const int tid  = threadIdx.x;
  const int wave = tid >> 5, lane = tid & 31;
  const int lm = lane & 15, lh = lane >> 4;

  v8f acc[DT];
#pragma unroll
  for (int t = 0; t < DT; ++t)
#pragma unroll
    for (int r = 0; r < 8; ++r) acc[t][r] = 0.f;

  for (int kb = 0; kb < F; kb += 32) {
    __syncthreads();
    {   // stage X[rb..+64][kb..+32] -> bf16 LDS; thread = half a row (16 f32)
      int row = tid >> 1, c0 = (tid & 1) * 16;
      const float* xp = X + (size_t)(rb + row) * F + kb + c0;
      bf16_t* sp = sX + row * 32 + c0;
#pragma unroll
      for (int q = 0; q < 16; q += 2)
        *(v2bf*)(sp + q) = pk2bf(xp[q], xp[q + 1]);
    }
    __syncthreads();

    // A fragment: 16x32 bf16, lane = row (M=lm), K = lh*16 + 2*v + p
    v16bf afr;
#pragma unroll
    for (int v = 0; v < 8; ++v) {
      int K = (lh << 4) + (v << 1);
      afr[2 * v]     = sX[(wave * 16 + lm) * 32 + K];
      afr[2 * v + 1] = sX[(wave * 16 + lm) * 32 + K + 1];
    }
#pragma unroll
    for (int t = 0; t < DT; ++t) {
      // B fragment: lane n = t*16+lm, K = lh*16 + 0..15 -> 32 contiguous bytes
      const bf16_t* bp = WT + ((size_t)(h * DT * 16 + t * 16 + lm)) * F + kb + (lh << 4);
      v16bf bfr = *(const v16bf*)bp;
      acc[t] = __builtin_amdgcn_wmma_f32_16x16x32_bf16(false, afr, false, bfr,
                                                       (short)0, acc[t], false, false);
    }
  }

  // C layout: lane=(col lm), VGPR r -> row j = r + 8*lh (+base); the 8 rows are
  // CONTIGUOUS in WhT's transposed layout -> one packed cvt + one 16B store.
#pragma unroll
  for (int t = 0; t < DT; ++t) {
    int n = t * 16 + lm;
    int j0 = rb + wave * 16 + (lh << 3);
    v8bf o = __builtin_convertvector(acc[t], v8bf);
    *(v8bf*)(WhT + ((size_t)(h * DT * 16 + n)) * N + j0) = o;
  }
}

// ---------------------------------------------------------------------------
// 3) s[h][n] = Wh[h][n]·a_src[h],  t likewise
// ---------------------------------------------------------------------------
__global__ void scores_kernel(const bf16_t* __restrict__ WhT,
                              const float* __restrict__ a_src,
                              const float* __restrict__ a_dst,
                              float* __restrict__ s, float* __restrict__ t, int d) {
  int idx = blockIdx.x * 256 + threadIdx.x;        // h*N + n
  if (idx >= HEADS * N) return;
  int n = idx & (N - 1), h = idx >> 12;
  float ss = 0.f, tt = 0.f;
  for (int dd = 0; dd < d; ++dd) {
    float w = bf2f(WhT[((size_t)(h * d + dd)) * N + n]);   // coalesced over n
    ss += w * a_src[h * d + dd];
    tt += w * a_dst[h * d + dd];
  }
  s[idx] = ss;
  t[idx] = tt;
}

// ---------------------------------------------------------------------------
// 4) per-row masked softmax stats: m = rowmax, invl = 1/sum(exp).
//    grid (H, N/8), block 256 (8 waves); one wave32 per row, shfl_xor reduce.
// ---------------------------------------------------------------------------
__global__ void softmax_stats(const unsigned char* __restrict__ adj,
                              const float* __restrict__ s,
                              const float* __restrict__ t,
                              float* __restrict__ mrow, float* __restrict__ invl) {
  int h = blockIdx.x;
  int i = blockIdx.y * 8 + (threadIdx.x >> 5);
  int lane = threadIdx.x & 31;
  float si = s[h * N + i];
  float m = -3.0e38f, l = 0.f;
  const unsigned char* arow = adj + (size_t)i * N;
  const float* th = t + h * N;
  for (int jb = 0; jb < N; jb += 128) {
    int j0 = jb + lane * 4;
    __builtin_prefetch(arow + j0 + 512, 0, 1);     // global_prefetch_b8
    unsigned a4 = *(const unsigned*)(arow + j0);
    float4 t4 = *(const float4*)(th + j0);
    float tv[4] = {t4.x, t4.y, t4.z, t4.w};
#pragma unroll
    for (int q = 0; q < 4; ++q) {
      if ((a4 >> (8 * q)) & 0xffu) {
        float e = si + tv[q];
        e = e > 0.f ? e : ALPHA * e;
        if (e > m) { l *= __expf(m - e); m = e; }
        l += __expf(e - m);
      }
    }
  }
#pragma unroll
  for (int off = 16; off; off >>= 1) {
    float mo = __shfl_xor(m, off, 32);
    float lo = __shfl_xor(l, off, 32);
    float mn = fmaxf(m, mo);
    l = l * __expf(m - mn) + lo * __expf(mo - mn);
    m = mn;
  }
  if (lane == 0) {
    mrow[h * N + i] = m;
    invl[h * N + i] = 1.f / l;
  }
}

// ---------------------------------------------------------------------------
// 5) fused attention aggregation: out = softmax(mask(leaky(s_i+t_j))) @ Wh.
//    Double-buffered async global->LDS staging overlapped with WMMA compute.
//    grid (N/64, H), block 128 (4 waves of 16 rows each).
// ---------------------------------------------------------------------------
template <int DT>   // DT = d/16
__global__ __launch_bounds__(128) void attn_kernel(const unsigned char* __restrict__ adj,
                                                   const bf16_t* __restrict__ WhT,
                                                   const float* __restrict__ s,
                                                   const float* __restrict__ t,
                                                   const float* __restrict__ mrow,
                                                   const float* __restrict__ invl,
                                                   float* __restrict__ out,
                                                   size_t head_off, size_t rstride,
                                                   int do_elu) {
  __shared__ __align__(32) bf16_t sWh[2][DT * 16 * 32];   // [buf][n][32 j]
  __shared__ __align__(16) unsigned char sAdj[2][64 * 32];
  __shared__ float sT[2][32];

  const int h   = blockIdx.y;
  const int rb  = blockIdx.x * 64;
  const int tid = threadIdx.x;
  const int wave = tid >> 5, lane = tid & 31;
  const int lm = lane & 15, lh = lane >> 4;

  const int i = rb + wave * 16 + lm;                   // A-fragment row
  const float si = s[h * N + i];
  const float mi = mrow[h * N + i];
  const float li = invl[h * N + i];

  v8f acc[DT];
#pragma unroll
  for (int tn = 0; tn < DT; ++tn)
#pragma unroll
    for (int r = 0; r < 8; ++r) acc[tn][r] = 0.f;

  // stage one 32-column tile (Wh slab + adjacency + t) into LDS buffer `buf`
  auto stage = [&](int jb, int buf) {
#pragma unroll
    for (int e = tid * 16; e < DT * 16 * 32; e += 128 * 16) {
      int nrow = e >> 5, c = e & 31;
      async_copy16(&sWh[buf][e],
                   WhT + ((size_t)(h * DT * 16 + nrow)) * N + jb + c);
    }
    int r = tid >> 1, c2 = (tid & 1) * 16;
    async_copy16(&sAdj[buf][r * 32 + c2], adj + (size_t)(rb + r) * N + jb + c2);
    if (tid < 32) sT[buf][tid] = t[h * N + jb + tid];
  };

  stage(0, 0);
  int cur = 0;
  for (int jb = 0; jb < N; jb += 32) {
    async_wait0();                 // current tile's DMA has landed in LDS
    __syncthreads();               // ... and everyone is done with the other buf
    if (jb + 32 < N) stage(jb + 32, cur ^ 1);   // prefetch next tile (overlaps)

    // A fragment: p_ij = adj ? exp(leaky(s_i+t_j)-m_i)/l_i : 0, in bf16
    v16bf afr;
#pragma unroll
    for (int v = 0; v < 8; ++v) {
      float pv[2];
#pragma unroll
      for (int p = 0; p < 2; ++p) {
        int K = (lh << 4) + (v << 1) + p;
        unsigned char a = sAdj[cur][(wave * 16 + lm) * 32 + K];
        float e = si + sT[cur][K];
        e = e > 0.f ? e : ALPHA * e;
        pv[p] = a ? __expf(e - mi) * li : 0.f;
      }
      v2bf pk = pk2bf(pv[0], pv[1]);
      afr[2 * v]     = pk[0];
      afr[2 * v + 1] = pk[1];
    }
#pragma unroll
    for (int tn = 0; tn < DT; ++tn) {
      v16bf bfr = *(const v16bf*)(&sWh[cur][(tn * 16 + lm) * 32 + (lh << 4)]);
      acc[tn] = __builtin_amdgcn_wmma_f32_16x16x32_bf16(false, afr, false, bfr,
                                                        (short)0, acc[tn], false, false);
    }
    cur ^= 1;
  }

  // epilogue: C layout -> out[h*head_off + row*rstride + n], optional ELU
#pragma unroll
  for (int tn = 0; tn < DT; ++tn)
#pragma unroll
    for (int r = 0; r < 8; ++r) {
      int n = tn * 16 + lm;
      int row = rb + wave * 16 + (lh << 3) + r;
      float v = acc[tn][r];
      if (do_elu) v = v > 0.f ? v : (__expf(v) - 1.f);
      out[(size_t)h * head_off + (size_t)row * rstride + n] = v;
    }
}

// ---------------------------------------------------------------------------
// 6) final layer: mean over heads
// ---------------------------------------------------------------------------
__global__ void mean_heads(const float* __restrict__ tmp, float* __restrict__ out) {
  int idx = blockIdx.x * 256 + threadIdx.x;
  if (idx >= N * OUT) return;
  float a = 0.f;
#pragma unroll
  for (int h = 0; h < HEADS; ++h) a += tmp[(size_t)h * N * OUT + idx];
  out[idx] = 0.25f * a;
}

// ---------------------------------------------------------------------------
// host orchestration
// ---------------------------------------------------------------------------
extern "C" void kernel_launch(void* const* d_in, const int* in_sizes, int n_in,
                              void* d_out, int out_size, void* d_ws, size_t ws_size,
                              hipStream_t stream) {
  (void)in_sizes; (void)n_in; (void)out_size; (void)ws_size;
  const float*         x   = (const float*)d_in[0];
  const unsigned char* adj = (const unsigned char*)d_in[1];   // jnp.bool_: 1 byte
  const float* Wl[3]  = {(const float*)d_in[2], (const float*)d_in[5], (const float*)d_in[8]};
  const float* asl[3] = {(const float*)d_in[3], (const float*)d_in[6], (const float*)d_in[9]};
  const float* adl[3] = {(const float*)d_in[4], (const float*)d_in[7], (const float*)d_in[10]};
  float* out = (float*)d_out;

  char* ws = (char*)d_ws;
  size_t off = 0;
  auto alloc = [&](size_t bytes) {
    void* p = ws + off;
    off += (bytes + 255) & ~(size_t)255;
    return p;
  };
  bf16_t* WT   = (bf16_t*)alloc((size_t)HEADS * 128 * F * sizeof(bf16_t));
  bf16_t* WhT  = (bf16_t*)alloc((size_t)HEADS * 128 * N * sizeof(bf16_t));
  float*  sbuf = (float*)alloc((size_t)HEADS * N * sizeof(float));
  float*  tbuf = (float*)alloc((size_t)HEADS * N * sizeof(float));
  float*  mbuf = (float*)alloc((size_t)HEADS * N * sizeof(float));
  float*  lbuf = (float*)alloc((size_t)HEADS * N * sizeof(float));
  float*  X1   = (float*)alloc((size_t)N * F * sizeof(float));
  float*  X2   = (float*)alloc((size_t)N * F * sizeof(float));
  float*  tmp  = (float*)alloc((size_t)HEADS * N * OUT * sizeof(float));

  auto run_layer = [&](const float* Xin, int layer, int d, float* outp,
                       size_t head_off, size_t rstride, int do_elu) {
    int wt_elems = HEADS * d * F;
    wt_kernel<<<(wt_elems + 255) / 256, 256, 0, stream>>>(Wl[layer], WT, d);
    if (d == 64)
      gemm_wh<4><<<dim3(N / 64, HEADS), 128, 0, stream>>>(Xin, WT, WhT);
    else
      gemm_wh<8><<<dim3(N / 64, HEADS), 128, 0, stream>>>(Xin, WT, WhT);
    scores_kernel<<<(HEADS * N + 255) / 256, 256, 0, stream>>>(WhT, asl[layer], adl[layer],
                                                               sbuf, tbuf, d);
    softmax_stats<<<dim3(HEADS, N / 8), 256, 0, stream>>>(adj, sbuf, tbuf, mbuf, lbuf);
    if (d == 64)
      attn_kernel<4><<<dim3(N / 64, HEADS), 128, 0, stream>>>(adj, WhT, sbuf, tbuf, mbuf,
                                                              lbuf, outp, head_off, rstride, do_elu);
    else
      attn_kernel<8><<<dim3(N / 64, HEADS), 128, 0, stream>>>(adj, WhT, sbuf, tbuf, mbuf,
                                                              lbuf, outp, head_off, rstride, do_elu);
  };

  run_layer(x,  0, 64,  X1,  (size_t)64,        (size_t)F,   1);  // concat + ELU
  run_layer(X1, 1, 64,  X2,  (size_t)64,        (size_t)F,   1);  // concat + ELU
  run_layer(X2, 2, 128, tmp, (size_t)N * OUT,   (size_t)OUT, 0);  // per-head -> tmp
  mean_heads<<<(N * OUT + 255) / 256, 256, 0, stream>>>(tmp, out);
}
